// MultiHeadAttention_29145648070704
// MI455X (gfx1250) — matrix operand order repaired and checked by reading, not compiled, not run on hardware
//
#include <hip/hip_runtime.h>
#include <hip/hip_bf16.h>

// ---------------------------------------------------------------------------
// MHA forward for MI455X (gfx1250): bf16 WMMA, f32 accumulate.
// B=2, S=2048, E=1024, H=16, D=64.
// attn_prob (537 MB) is streamed LDS -> HBM with global_store_async_from_lds.
// ---------------------------------------------------------------------------

#define Bsz 2
#define Ssz 2048
#define Esz 1024
#define Hsz 16
#define Dsz 64
#define Msz (Bsz * Ssz)   // 4096 rows of activations

typedef __bf16  bf16;
typedef __attribute__((ext_vector_type(16))) __bf16 v16bf;
typedef __attribute__((ext_vector_type(8)))  __bf16 bf16x8;
typedef __attribute__((ext_vector_type(4)))  __bf16 bf16x4;
typedef __attribute__((ext_vector_type(8)))  float  v8f;
typedef __attribute__((ext_vector_type(4)))  float  f32x4;

// A-fragment (16x32, 16-bit): lane<16 -> row=lane, K = {0..7, 16..23};
// lane>=16 -> row=lane-16, K = {8..15, 24..31}. Caller pre-offsets by koffA.
__device__ __forceinline__ v16bf load_frag_a_bf16(const bf16* p) {
  bf16x8 lo = *(const bf16x8*)(p);
  bf16x8 hi = *(const bf16x8*)(p + 16);
  v16bf r;
#pragma unroll
  for (int i = 0; i < 8; ++i) { r[i] = lo[i]; r[i + 8] = hi[i]; }
  return r;
}

// Same A pattern but from f32 memory (global or LDS), converting to bf16.
__device__ __forceinline__ v16bf load_frag_a_f32(const float* p) {
  f32x4 f0 = *(const f32x4*)(p);
  f32x4 f1 = *(const f32x4*)(p + 4);
  f32x4 f2 = *(const f32x4*)(p + 16);
  f32x4 f3 = *(const f32x4*)(p + 20);
  v16bf r;
#pragma unroll
  for (int i = 0; i < 4; ++i) {
    r[i]      = (bf16)f0[i];
    r[i + 4]  = (bf16)f1[i];
    r[i + 8]  = (bf16)f2[i];
    r[i + 12] = (bf16)f3[i];
  }
  return r;
}

// B-fragment (32x16, 16-bit): lane<16 -> col=lane, K = {0..7, 16..23};
// lane>=16 -> col=lane-16, K = {8..15, 24..31}. Caller pre-offsets by koffB.
__device__ __forceinline__ v16bf load_frag_b_contig(const bf16* p) {
  bf16x8 lo = *(const bf16x8*)(p);
  bf16x8 hi = *(const bf16x8*)(p + 16);
  v16bf r;
#pragma unroll
  for (int i = 0; i < 8; ++i) { r[i] = lo[i]; r[i + 8] = hi[i]; }
  return r;
}

__device__ __forceinline__ v8f wmma_bf16(v16bf a, v16bf b, v8f c) {
  return __builtin_amdgcn_wmma_f32_16x16x32_bf16(false, a, false, b,
                                                 (short)0, c, false, false);
}

// ---------------------------------------------------------------------------
// Kernel 1: f32 -> bf16 convert (weights).
// ---------------------------------------------------------------------------
__global__ __launch_bounds__(256) void cvt_f32_bf16(const float* __restrict__ src,
                                                    bf16* __restrict__ dst, int n) {
  int i = (blockIdx.x * 256 + threadIdx.x) * 4;
  if (i + 3 < n) {
    f32x4 f = *(const f32x4*)(src + i);
    bf16x4 o;
#pragma unroll
    for (int j = 0; j < 4; ++j) o[j] = (bf16)f[j];
    *(bf16x4*)(dst + i) = o;
  }
}

// ---------------------------------------------------------------------------
// Kernel 2/4: projection GEMM  Y = X @ W^T + b  via WMMA bf16.
//   MODE 0: X is f32 [M,E] (converted on the fly), Y -> bf16 in [B,H,S,D]
//   MODE 1: X is bf16 [M,E], Y -> f32 [M,E] (final output projection)
// One wave computes one 16x16 tile; 8 waves (256 threads) per block.
// ---------------------------------------------------------------------------
template <int MODE>
__global__ __launch_bounds__(256) void proj_gemm(const void* __restrict__ Aptr,
                                                 const bf16* __restrict__ W,
                                                 const float* __restrict__ bias,
                                                 void* __restrict__ outp) {
  const int lane = threadIdx.x & 31;
  const int wid  = threadIdx.x >> 5;
  const int tile = blockIdx.x * 8 + wid;
  const int tm = tile >> 6;        // M/16 = 256 tiles, E/16 = 64 tiles
  const int tn = tile & 63;
  const int l15   = lane & 15;
  const int koffA = (lane < 16) ? 0 : 8;
  const int koffB = (lane < 16) ? 0 : 8;
  const int mrow  = tm * 16 + l15;
  const int ncol  = tn * 16 + l15;

  const bf16*  Ab = (const bf16*)Aptr;
  const float* Af = (const float*)Aptr;
  const bf16*  wrow = W + (size_t)ncol * Esz + koffB;

  v8f acc = {};
#pragma unroll 4
  for (int k0 = 0; k0 < Esz; k0 += 32) {
    v16bf a;
    if (MODE == 0)
      a = load_frag_a_f32(Af + (size_t)mrow * Esz + k0 + koffA);
    else
      a = load_frag_a_bf16(Ab + (size_t)mrow * Esz + k0 + koffA);
    v16bf b = load_frag_b_contig(wrow + k0);
    acc = wmma_bf16(a, b, acc);
  }

  const float bv = bias[ncol];
  const int rbase = (lane < 16) ? 0 : 8;
#pragma unroll
  for (int r = 0; r < 8; ++r) {
    const int m  = tm * 16 + r + rbase;
    const float val = acc[r] + bv;
    if (MODE == 0) {
      const int b_ = m >> 11, s = m & (Ssz - 1);
      const int h  = ncol >> 6, d = ncol & (Dsz - 1);
      ((bf16*)outp)[(((size_t)(b_ * Hsz + h)) * Ssz + s) * Dsz + d] = (bf16)val;
    } else {
      ((float*)outp)[(size_t)m * Esz + ncol] = val;
    }
  }
}

// ---------------------------------------------------------------------------
// Kernel 3: fused attention for one (b,h, 16-row q-tile).
// 128 threads = 4 waves.  LDS: 16x2048 f32 score strip (128 KB), reductions,
// 64x128 bf16 transposed V staging tile (16 KB).
// ---------------------------------------------------------------------------
#define KC 128   // keys staged per PV chunk

__global__ __launch_bounds__(128) void attn_kernel(const bf16* __restrict__ q,
                                                   const bf16* __restrict__ k,
                                                   const bf16* __restrict__ v,
                                                   float* __restrict__ attn_prob,
                                                   bf16* __restrict__ attn_out) {
  extern __shared__ char smem_raw[];
  float* sc   = (float*)smem_raw;        // [16][2048]  = 131072 B
  float* pmax = sc + 16 * 2048;          // [16][8]
  float* rmax = pmax + 128;              // [16]
  float* psum = rmax + 16;               // [16][8]
  float* rsum = psum + 128;              // [16]
  bf16*  vT   = (bf16*)(rsum + 16);      // [64][KC] transposed V chunk, 16 KB

  const int lane = threadIdx.x & 31;
  const int wid  = threadIdx.x >> 5;     // 0..3
  const int l15  = lane & 15;
  const int koffA = (lane < 16) ? 0 : 8;
  const int koffB = (lane < 16) ? 0 : 8;

  const int bh    = blockIdx.x >> 7;     // S/16 = 128 q-tiles per (b,h)
  const int qt    = blockIdx.x & 127;
  const int qbase = qt * 16;

  // ---- Q fragments for d=0..31 and d=32..63 --------------------------------
  const bf16* qp = q + ((size_t)bh * Ssz + qbase + l15) * Dsz;
  const v16bf aq0 = load_frag_a_bf16(qp + koffA);
  const v16bf aq1 = load_frag_a_bf16(qp + 32 + koffA);

  // ---- Phase 1: scores = Q K^T / sqrt(D), causal mask, into LDS ------------
#pragma unroll 2
  for (int kt = wid; kt < Ssz / 16; kt += 4) {
    const bf16* kp = k + ((size_t)bh * Ssz + kt * 16 + l15) * Dsz + koffB;
    v16bf b0 = load_frag_b_contig(kp);
    v16bf b1 = load_frag_b_contig(kp + 32);
    v8f s8 = {};
    s8 = wmma_bf16(aq0, b0, s8);
    s8 = wmma_bf16(aq1, b1, s8);
    const int rb = (lane < 16) ? 0 : 8;
#pragma unroll
    for (int r = 0; r < 8; ++r) {
      const int row = r + rb;                 // 0..15 local q row
      const int kg  = kt * 16 + l15;          // global key index
      float val = s8[r] * 0.125f;             // 1/sqrt(64)
      if (kg > qbase + row) val = -__builtin_inff();
      sc[row * Ssz + kg] = val;
    }
  }
  __syncthreads();

  // ---- Phase 2: softmax over each 2048-wide row (8 threads per row) --------
  const int row = threadIdx.x >> 3;
  const int seg = threadIdx.x & 7;
  float* rp = sc + row * Ssz + seg * 256;     // this thread's 256-col segment
  {
    float mx = -__builtin_inff();
    for (int c = 0; c < 256; c += 4) {
      f32x4 f = *(const f32x4*)(rp + c);
      mx = fmaxf(mx, fmaxf(fmaxf(f[0], f[1]), fmaxf(f[2], f[3])));
    }
    pmax[row * 8 + seg] = mx;
  }
  __syncthreads();
  if (seg == 0) {
    float m2 = pmax[row * 8];
#pragma unroll
    for (int j = 1; j < 8; ++j) m2 = fmaxf(m2, pmax[row * 8 + j]);
    rmax[row] = m2;
  }
  __syncthreads();
  {
    const float m = rmax[row];
    float s = 0.f;
    for (int c = 0; c < 256; c += 4) {
      f32x4 f = *(const f32x4*)(rp + c);
      f32x4 e;
#pragma unroll
      for (int j = 0; j < 4; ++j) { e[j] = __expf(f[j] - m); s += e[j]; }
      *(f32x4*)(rp + c) = e;
    }
    psum[row * 8 + seg] = s;
  }
  __syncthreads();
  if (seg == 0) {
    float s2 = psum[row * 8];
#pragma unroll
    for (int j = 1; j < 8; ++j) s2 += psum[row * 8 + j];
    rsum[row] = s2;
  }
  __syncthreads();
  {
    // normalize in LDS (vectorized), then DMA the finished 1 KB segment
    // straight from LDS to HBM with async stores (ASYNCcnt) so the 537 MB
    // attn_prob stream overlaps the PV WMMA phase below.
    const float inv = 1.0f / rsum[row];
    for (int c = 0; c < 256; c += 4) {
      f32x4 f = *(const f32x4*)(rp + c);
#pragma unroll
      for (int j = 0; j < 4; ++j) f[j] *= inv;
      *(f32x4*)(rp + c) = f;
    }
    const float* gseg = attn_prob + ((size_t)bh * Ssz + qbase + row) * Ssz + seg * 256;
    unsigned long long gaddr = (unsigned long long)(uintptr_t)gseg;
    unsigned int       laddr = (unsigned int)(uintptr_t)rp;
    asm volatile("s_wait_dscnt 0x0" ::: "memory");  // LDS data visible to DMA
    for (int c = 0; c < 256; c += 4) {
      asm volatile("global_store_async_from_lds_b128 %0, %1, off"
                   :: "v"(gaddr + (unsigned long long)c * 4ull),
                      "v"(laddr + (unsigned int)c * 4u)
                   : "memory");
    }
  }
  __syncthreads();

  // ---- Phase 3: out = P @ V. Wave w owns d-columns [w*16, w*16+16). --------
  v8f acco = {};
  const int klb  = threadIdx.x >> 2;        // 0..31: base V row within chunk
  const int dseg = (threadIdx.x & 3) * 16;  // d sub-range this thread stages
  for (int kc = 0; kc < Ssz; kc += KC) {
    // stage V[kc..kc+KC) x 64d transposed into vT[d][kloc]
#pragma unroll
    for (int sub = 0; sub < KC / 32; ++sub) {
      const int kl = klb + sub * 32;
      const bf16* vp = v + ((size_t)bh * Ssz + kc + kl) * Dsz + dseg;
      bf16x8 v0 = *(const bf16x8*)vp;
      bf16x8 v1 = *(const bf16x8*)(vp + 8);
#pragma unroll
      for (int i = 0; i < 8; ++i) {
        vT[(dseg + i) * KC + kl]     = v0[i];
        vT[(dseg + 8 + i) * KC + kl] = v1[i];
      }
    }
    __syncthreads();
#pragma unroll
    for (int j = 0; j < KC / 32; ++j) {
      // A: probability tile (f32 in LDS -> bf16 fragment)
      v16bf pa = load_frag_a_f32(sc + (size_t)l15 * Ssz + kc + 32 * j + koffA);
      // B: transposed V tile, contiguous in LDS
      v16bf pb = load_frag_b_contig(vT + (wid * 16 + l15) * KC + 32 * j + koffB);
      acco = wmma_bf16(pa, pb, acco);
    }
    __syncthreads();
  }

  // merge heads -> bf16 [B, S, E]
  const int b_ = bh >> 4;
  const int h  = bh & 15;
  const int rbase = (lane < 16) ? 0 : 8;
#pragma unroll
  for (int r = 0; r < 8; ++r) {
    const int mr = qbase + r + rbase;
    const int nd = wid * 16 + l15;
    attn_out[((size_t)b_ * Ssz + mr) * Esz + h * Dsz + nd] = (bf16)acco[r];
  }
}

// ---------------------------------------------------------------------------
// Host launcher
// ---------------------------------------------------------------------------
extern "C" void kernel_launch(void* const* d_in, const int* in_sizes, int n_in,
                              void* d_out, int out_size, void* d_ws, size_t ws_size,
                              hipStream_t stream) {
  (void)in_sizes; (void)n_in; (void)out_size; (void)ws_size;

  const float* Q   = (const float*)d_in[0];
  const float* K   = (const float*)d_in[1];
  const float* V   = (const float*)d_in[2];
  /* d_in[3] = mask (bool) -- causal mask computed analytically */
  const float* W_q = (const float*)d_in[4];
  const float* b_q = (const float*)d_in[5];
  const float* W_k = (const float*)d_in[6];
  const float* b_k = (const float*)d_in[7];
  const float* W_v = (const float*)d_in[8];
  const float* b_v = (const float*)d_in[9];
  const float* W_o = (const float*)d_in[10];
  const float* b_o = (const float*)d_in[11];

  // Workspace layout (bf16 elements), ~40 MB total.
  bf16* ws = (bf16*)d_ws;
  const size_t WELEM = (size_t)Esz * Esz;             // 1 Mi elems per weight
  const size_t QKVE  = (size_t)Bsz * Hsz * Ssz * Dsz; // 4 Mi elems
  bf16* Wqb = ws;
  bf16* Wkb = Wqb + WELEM;
  bf16* Wvb = Wkb + WELEM;
  bf16* Wob = Wvb + WELEM;
  bf16* qb  = Wob + WELEM;
  bf16* kb  = qb + QKVE;
  bf16* vb  = kb + QKVE;
  bf16* ao  = vb + QKVE;                              // attn_out, [B,S,E] bf16

  float* out_proj  = (float*)d_out;                              // [B,S,E]
  float* attn_prob = out_proj + (size_t)Bsz * Ssz * Esz;         // [B,H,S,S]

  // 1) weights -> bf16
  {
    const int n = (int)WELEM;
    const int grid = n / (256 * 4);
    cvt_f32_bf16<<<grid, 256, 0, stream>>>(W_q, Wqb, n);
    cvt_f32_bf16<<<grid, 256, 0, stream>>>(W_k, Wkb, n);
    cvt_f32_bf16<<<grid, 256, 0, stream>>>(W_v, Wvb, n);
    cvt_f32_bf16<<<grid, 256, 0, stream>>>(W_o, Wob, n);
  }

  // 2) q/k/v projections: [4096,1024] x [1024,1024]^T, 16384 tiles / 8 waves
  const int proj_grid = (Msz / 16) * (Esz / 16) / 8;  // 2048
  proj_gemm<0><<<proj_grid, 256, 0, stream>>>(Q, Wqb, b_q, qb);
  proj_gemm<0><<<proj_grid, 256, 0, stream>>>(K, Wkb, b_k, kb);
  proj_gemm<0><<<proj_grid, 256, 0, stream>>>(V, Wvb, b_v, vb);

  // 3) fused attention
  constexpr size_t SMEM = (size_t)16 * Ssz * 4            // score strip
                        + (128 + 16 + 128 + 16) * 4       // reductions
                        + (size_t)64 * KC * 2;            // vT
  hipFuncSetAttribute((const void*)attn_kernel,
                      hipFuncAttributeMaxDynamicSharedMemorySize, (int)SMEM);
  const int attn_grid = Bsz * Hsz * (Ssz / 16);  // 4096
  attn_kernel<<<attn_grid, 128, SMEM, stream>>>(qb, kb, vb, attn_prob, ao);

  // 4) output projection -> f32
  proj_gemm<1><<<proj_grid, 256, 0, stream>>>(ao, Wob, b_o, out_proj);
}
